// network_representation_module_GIN_57346403336484
// MI455X (gfx1250) — matrix operand order repaired and checked by reading, not compiled, not run on hardware
//
#include <hip/hip_runtime.h>

// ---------------------------------------------------------------------------
// GIN 2-layer forward for MI455X (gfx1250, wave32).
//
// Cost model: GEMMs are ~3.3 GFLOP total (negligible); the scatter-sum over
// 1.6M edges dominates.  All buffers (features/agg/hidden ~77MB + 12.8MB edge
// list) fit in the 192MB L2, so gathers + f32 atomics run at L2 speed, far
// above the 23.3 TB/s HBM roof.  GEMM uses v_wmma_f32_16x16x32_f16 with
// in-register f16 A fragments and pre-packed fragment-ordered f16 B.
// ---------------------------------------------------------------------------

typedef __attribute__((ext_vector_type(16))) _Float16 v16h;
typedef __attribute__((ext_vector_type(8)))  float    v8f;

#define D_HID 128

// ---------------------------------------------------------------------------
// Pack W (128x128 row-major f32, W[k][n]) into WMMA B-fragment order (f16).
// Fragment (nt, ks): lane l<16 -> column n = nt*16+l, element v holds K=ks*32+{0..7,16..23}[v].
//                    lane l>=16 -> column n = nt*16+l-16, element v holds K=ks*32+{8..15,24..31}[v].
// Linear layout: Wp[(((nt*4+ks)*32 + lane)*16 + v]  -> 32B contiguous per lane.
// Total: 8*4*32*16 = 16384 halves = 32KB per weight matrix.
// ---------------------------------------------------------------------------
__global__ void pack_w_kernel(const float* __restrict__ W, _Float16* __restrict__ Wp) {
    int idx = blockIdx.x * blockDim.x + threadIdx.x;   // 0..16383
    if (idx >= 8 * 4 * 32 * 16) return;
    int v    = idx & 15;
    int lane = (idx >> 4) & 31;
    int ks   = (idx >> 9) & 3;
    int nt   = (idx >> 11) & 7;
    int kk   = ((v < 8) ? v : (v + 8)) + ((lane < 16) ? 0 : 8);
    int nn   = nt * 16 + (lane & 15);
    int K    = ks * 32 + kk;
    Wp[idx] = (_Float16)W[K * D_HID + nn];
}

// ---------------------------------------------------------------------------
// Edge scatter: one wave per edge.  Lane l gathers float4 l of x[src] (512B
// per edge, coalesced) and atomically adds into agg[dst].  agg is L2-resident.
// ---------------------------------------------------------------------------
__global__ void scatter_add_kernel(const float* __restrict__ x,
                                   const int* __restrict__ src,
                                   const int* __restrict__ dst,
                                   float* __restrict__ agg, int nEdges) {
    int gt   = blockIdx.x * blockDim.x + threadIdx.x;
    int e    = gt >> 5;
    int lane = threadIdx.x & 31;
    if (e >= nEdges) return;
    int s = src[e];
    int d = dst[e];
    const float4* xs = (const float4*)(x + (size_t)s * D_HID);
    float4 val = xs[lane];
    float* ad = agg + (size_t)d * D_HID + lane * 4;
    atomicAdd(ad + 0, val.x);
    atomicAdd(ad + 1, val.y);
    atomicAdd(ad + 2, val.z);
    atomicAdd(ad + 3, val.w);
}

// ---------------------------------------------------------------------------
// GIN GEMM: out[m][n] = sum_k ((1+eps)*x[m][k] + agg[m][k]) * W[k][n] + b[n]
// One wave per 16-row tile; full N=128 per wave (8 accum tiles of 8 VGPRs).
// A fragment layout (16-bit A, 16x32): lanes 0-15 row m=lane hold
// K = {0..7} in v0..3 halves and {16..23} in v4..7; lanes 16-31 same row,
// K = {8..15} and {24..31}.  So each lane loads two contiguous 8-float runs.
// ---------------------------------------------------------------------------
__global__ void gin_gemm_kernel(const float* __restrict__ x,
                                const float* __restrict__ agg,
                                const float* __restrict__ epsp,
                                const _Float16* __restrict__ Wp,
                                const float* __restrict__ bias,
                                float* __restrict__ out, int nTiles) {
    int wave = (blockIdx.x * blockDim.x + threadIdx.x) >> 5;
    int lane = threadIdx.x & 31;
    if (wave >= nTiles) return;                 // uniform per wave -> EXEC stays all-1s

    float epsv = 1.0f + *epsp;                  // scalar (uniform) load
    int row  = wave * 16 + (lane & 15);
    int kb   = (lane < 16) ? 0 : 8;
    const float* xr = x   + (size_t)row * D_HID;
    const float* ar = agg + (size_t)row * D_HID;

    // Build 4 A fragments (K = 0..127 in steps of 32), converting to f16.
    v16h afrag[4];
#pragma unroll
    for (int ks = 0; ks < 4; ++ks) {
        int k0 = ks * 32 + kb;
#pragma unroll
        for (int v = 0; v < 8; ++v) {
            afrag[ks][v]     = (_Float16)(epsv * xr[k0 + v]      + ar[k0 + v]);
            afrag[ks][8 + v] = (_Float16)(epsv * xr[k0 + 16 + v] + ar[k0 + 16 + v]);
        }
    }

#pragma unroll
    for (int nt = 0; nt < 8; ++nt) {
        v8f c = {};
#pragma unroll
        for (int ks = 0; ks < 4; ++ks) {
            v16h b = *(const v16h*)(Wp + (((nt * 4 + ks) * 32 + lane) << 4));
            c = __builtin_amdgcn_wmma_f32_16x16x32_f16(
                    /*neg_a=*/false, afrag[ks], /*neg_b=*/false, b,
                    /*c_mod=*/(short)0, c, /*reuse_a=*/false, /*reuse_b=*/false);
        }
        // C/D layout: VGPR r: lanes 0-15 -> (M=r, N=lane); lanes 16-31 -> (M=8+r, N=lane-16)
        int col   = nt * 16 + (lane & 15);
        int rbase = wave * 16 + ((lane < 16) ? 0 : 8);
        float bv  = bias[col];
#pragma unroll
        for (int r = 0; r < 8; ++r)
            out[(size_t)(rbase + r) * D_HID + col] = c[r] + bv;
    }
}

// ---------------------------------------------------------------------------
// BatchNorm stats: per-channel sum & sumsq.  128 threads/block = one channel
// per thread; each block strides over rows (fully coalesced 512B row reads),
// then one atomicAdd per channel per block.
// ---------------------------------------------------------------------------
__global__ void bn_stats_kernel(const float* __restrict__ x,
                                float* __restrict__ sums,
                                float* __restrict__ sumsq, int M) {
    int c = threadIdx.x;                        // 0..127
    float s = 0.0f, s2 = 0.0f;
    for (int r = blockIdx.x; r < M; r += gridDim.x) {
        float v = x[(size_t)r * D_HID + c];
        s  += v;
        s2 += v * v;
    }
    atomicAdd(&sums[c], s);
    atomicAdd(&sumsq[c], s2);
}

__global__ void bn_final_kernel(const float* __restrict__ sums,
                                const float* __restrict__ sumsq,
                                const float* __restrict__ gamma,
                                const float* __restrict__ beta,
                                float* __restrict__ scale,
                                float* __restrict__ shift, float invM) {
    int c = threadIdx.x;                        // 0..127
    float mu  = sums[c] * invM;
    float var = sumsq[c] * invM - mu * mu;      // biased variance (matches ref)
    float sc  = gamma[c] * rsqrtf(var + 1e-5f);
    scale[c] = sc;
    shift[c] = beta[c] - mu * sc;
}

__global__ void bn_apply_relu_kernel(const float* __restrict__ x,
                                     const float* __restrict__ scale,
                                     const float* __restrict__ shift,
                                     float* __restrict__ y, int total) {
    int i = blockIdx.x * blockDim.x + threadIdx.x;
    if (i >= total) return;
    int c = i & (D_HID - 1);
    float v = fmaf(x[i], scale[c], shift[c]);
    y[i] = v > 0.0f ? v : 0.0f;
}

// ---------------------------------------------------------------------------
// Host-side orchestration (graph-capture safe: only async ops on `stream`).
// ---------------------------------------------------------------------------
extern "C" void kernel_launch(void* const* d_in, const int* in_sizes, int n_in,
                              void* d_out, int out_size, void* d_ws, size_t ws_size,
                              hipStream_t stream) {
    const int D = D_HID;
    const int M  = in_sizes[0] / D;        // 50000 nodes
    const int nE = in_sizes[1] / 2;        // 1,600,000 edges

    const float* feat  = (const float*)d_in[0];
    const int*   src   = (const int*)d_in[1];
    const int*   dst   = src + nE;
    const float* W1    = (const float*)d_in[2];
    const float* b1    = (const float*)d_in[3];
    const float* eps1  = (const float*)d_in[4];
    const float* g1    = (const float*)d_in[5];
    const float* be1   = (const float*)d_in[6];
    const float* W2    = (const float*)d_in[7];
    const float* b2    = (const float*)d_in[8];
    const float* eps2  = (const float*)d_in[9];
    const float* g2    = (const float*)d_in[10];
    const float* be2   = (const float*)d_in[11];

    // Workspace layout (all L2-resident: ~77 MB total)
    char* ws = (char*)d_ws;
    size_t nodeBytes = (size_t)M * D * sizeof(float);          // 25.6 MB
    float*    agg  = (float*)ws;
    float*    h    = (float*)(ws + nodeBytes);
    float*    y1   = (float*)(ws + 2 * nodeBytes);
    _Float16* Wp1  = (_Float16*)(ws + 3 * nodeBytes);          // 32 KB
    _Float16* Wp2  = Wp1 + 16384;                              // 32 KB
    float*    sums = (float*)(ws + 3 * nodeBytes + 65536);
    float*    sumsq = sums + 128;
    float*    scale = sums + 256;
    float*    shift = sums + 384;

    const int nTiles     = M / 16;                             // 3125 (M % 16 == 0)
    const int gemmBlocks = (nTiles * 32 + 255) / 256;
    const int scatBlocks = (int)(((long long)nE * 32 + 255) / 256);
    const int elemBlocks = (M * D + 255) / 256;
    const float invM = 1.0f / (float)M;

    // Pack both weight matrices into WMMA B-fragment order (once per call).
    pack_w_kernel<<<64, 256, 0, stream>>>(W1, Wp1);
    pack_w_kernel<<<64, 256, 0, stream>>>(W2, Wp2);

    // ---- Layer 1 ----
    hipMemsetAsync(agg, 0, nodeBytes, stream);
    hipMemsetAsync(sums, 0, 2 * 128 * sizeof(float), stream);
    scatter_add_kernel<<<scatBlocks, 256, 0, stream>>>(feat, src, dst, agg, nE);
    gin_gemm_kernel<<<gemmBlocks, 256, 0, stream>>>(feat, agg, eps1, Wp1, b1, h, nTiles);
    bn_stats_kernel<<<256, 128, 0, stream>>>(h, sums, sumsq, M);
    bn_final_kernel<<<1, 128, 0, stream>>>(sums, sumsq, g1, be1, scale, shift, invM);
    bn_apply_relu_kernel<<<elemBlocks, 256, 0, stream>>>(h, scale, shift, y1, M * D);

    // ---- Layer 2 ----
    hipMemsetAsync(agg, 0, nodeBytes, stream);
    hipMemsetAsync(sums, 0, 2 * 128 * sizeof(float), stream);
    scatter_add_kernel<<<scatBlocks, 256, 0, stream>>>(y1, src, dst, agg, nE);
    gin_gemm_kernel<<<gemmBlocks, 256, 0, stream>>>(y1, agg, eps2, Wp2, b2, h, nTiles);
    bn_stats_kernel<<<256, 128, 0, stream>>>(h, sums, sumsq, M);
    bn_final_kernel<<<1, 128, 0, stream>>>(sums, sumsq, g2, be2, scale, shift, invM);
    bn_apply_relu_kernel<<<elemBlocks, 256, 0, stream>>>(h, scale, shift, (float*)d_out, M * D);
}
